// MultiHeadAttention_2276332666927
// MI455X (gfx1250) — matrix operand order repaired and checked by reading, not compiled, not run on hardware
//
#include <hip/hip_runtime.h>
#include <math.h>

#define BB 8
#define NN 2048
#define HH 8
#define DD 64
#define DM 512

typedef __attribute__((ext_vector_type(16))) __bf16 v16bf;
typedef __attribute__((ext_vector_type(8)))  float  v8f;

union Frag16 { unsigned u[8]; v16bf v; };

static __device__ __forceinline__ unsigned short f2bf(float f) {
    union { __bf16 b; unsigned short u; } t;
    t.b = (__bf16)f;
    return t.u;
}
static __device__ __forceinline__ unsigned pack2(float lo, float hi) {
    union { __bf16 b[2]; unsigned u; } t;
    t.b[0] = (__bf16)lo;
    t.b[1] = (__bf16)hi;
    return t.u;
}

// Per-lane async copy of 16 bytes global -> LDS (ASYNCcnt-tracked).
static __device__ __forceinline__ void async_cp16(unsigned lds_off, const void* gptr) {
    unsigned long long g = (unsigned long long)(uintptr_t)gptr;
    asm volatile("global_load_async_to_lds_b128 %0, %1, off"
                 :: "v"(lds_off), "v"(g) : "memory");
}
// Compiler-level LDS ordering fence (HW keeps per-wave DS ops in order).
static __device__ __forceinline__ void wave_lds_fence() {
    asm volatile("" ::: "memory");
}

// ---------------------------------------------------------------------------
// Kernel 1: per-head QKV projection. One wave = one 16-row tile of one (b,h).
// Writes Q,K row-major bf16 [bh][n][d]; V transposed bf16 [bh][d][n].
// ---------------------------------------------------------------------------
__global__ __launch_bounds__(256) void qkv_proj_kernel(
    const float* __restrict__ x,
    const float* __restrict__ Wq, const float* __restrict__ bq,
    const float* __restrict__ Wk, const float* __restrict__ bk,
    const float* __restrict__ Wv, const float* __restrict__ bv,
    unsigned short* __restrict__ qb, unsigned short* __restrict__ kb,
    unsigned short* __restrict__ vt)
{
    const int lane = threadIdx.x & 31;
    const int wave = threadIdx.x >> 5;
    const int tile = blockIdx.x * 8 + wave;          // B*H*(N/16) tiles
    const int TPB  = NN / 16;
    const int bh   = tile / TPB;
    const int nblk = tile % TPB;
    const int h    = bh % HH;
    const int b    = bh / HH;

    const int nidx = lane & 15;
    const int half = lane >> 4;

    // A fragments: X tile 16x64 (two 16x32 chunks), fp32 -> bf16
    const float* xrow = x + ((size_t)b * NN + nblk * 16 + nidx) * DM + h * DD;
    Frag16 a[2];
#pragma unroll
    for (int cc = 0; cc < 2; ++cc)
#pragma unroll
        for (int j = 0; j < 8; ++j) {
            int k = cc * 32 + half * 16 + (j << 1);
            a[cc].u[j] = pack2(xrow[k], xrow[k + 1]);
        }

    const float* Ws[3] = { Wq, Wk, Wv };
    const float* bs[3] = { bq, bk, bv };

#pragma unroll
    for (int w = 0; w < 3; ++w) {
        const float* Wh = Ws[w] + (size_t)h * DD * DD;
#pragma unroll
        for (int colc = 0; colc < 4; ++colc) {
            const int col = colc * 16 + nidx;          // output feature (B-frag N)
            const float* wrow = Wh + (size_t)col * DD; // W[h][col][k] contiguous in k
            Frag16 bw[2];
#pragma unroll
            for (int cc = 0; cc < 2; ++cc)
#pragma unroll
                for (int j = 0; j < 8; ++j) {
                    int k = cc * 32 + half * 16 + (j << 1);
                    bw[cc].u[j] = pack2(wrow[k], wrow[k + 1]);
                }
            v8f acc = {};
            acc = __builtin_amdgcn_wmma_f32_16x16x32_bf16(false, a[0].v, false, bw[0].v,
                                                          (short)0, acc, false, false);
            acc = __builtin_amdgcn_wmma_f32_16x16x32_bf16(false, a[1].v, false, bw[1].v,
                                                          (short)0, acc, false, false);
            const float bias = bs[w][h * DD + col];
#pragma unroll
            for (int r = 0; r < 8; ++r) {
                const int row = nblk * 16 + r + half * 8;
                const unsigned short val = f2bf(acc[r] + bias);
                if (w == 0)
                    qb[((size_t)bh * NN + row) * DD + col] = val;
                else if (w == 1)
                    kb[((size_t)bh * NN + row) * DD + col] = val;
                else
                    vt[((size_t)bh * DD + col) * NN + row] = val;  // V transposed
            }
        }
    }
}

// ---------------------------------------------------------------------------
// Kernel 2: flash attention. Block = 4 waves sharing one (b,h); each wave owns
// 16 query rows. K/V tiles (32 keys) are double-buffered in LDS and filled by
// async global->LDS copies overlapped with compute on the previous tile.
// ---------------------------------------------------------------------------
__global__ __launch_bounds__(128) void flash_attn_kernel(
    const unsigned short* __restrict__ qb,
    const unsigned short* __restrict__ kb,
    const unsigned short* __restrict__ vt,
    float* __restrict__ o)
{
    __shared__ unsigned short Kl[2][32][64];   // [buf][key][d]      8 KB
    __shared__ unsigned short Vl[2][64][32];   // [buf][d][key]      8 KB
    __shared__ unsigned short Pl[4][16][32];   // per-wave P staging 4 KB

    const int tid  = threadIdx.x;
    const int lane = tid & 31;
    const int wave = tid >> 5;
    const int QPB  = NN / 64;                  // query blocks per (b,h)
    const int bh   = blockIdx.x / QPB;
    const int qblk = blockIdx.x % QPB;
    const int qbase = qblk * 64 + wave * 16;

    const int nidx = lane & 15;
    const int half = lane >> 4;

    const unsigned short* kbh = kb + (size_t)bh * NN * DD;
    const unsigned short* vbh = vt + (size_t)bh * DD * NN;

    // Q A-fragments (persist across the key loop)
    const unsigned short* qrow = qb + ((size_t)bh * NN + qbase + nidx) * DD;
    Frag16 qa[2];
#pragma unroll
    for (int cc = 0; cc < 2; ++cc)
#pragma unroll
        for (int j = 0; j < 8; ++j) {
            int k = cc * 32 + half * 16 + (j << 1);
            qa[cc].u[j] = *(const unsigned*)(qrow + k);
        }

    v8f oacc[4];
    float m[8], l[8];
#pragma unroll
    for (int c = 0; c < 4; ++c) oacc[c] = v8f{};
#pragma unroll
    for (int r = 0; r < 8; ++r) { m[r] = -3.0e38f; l[r] = 0.0f; }

    const float scale = 0.125f;  // 1/sqrt(64)

    // LDS byte offsets of the staging buffers (workgroup-relative)
    const unsigned klOff[2] = { (unsigned)(uintptr_t)&Kl[0][0][0],
                                (unsigned)(uintptr_t)&Kl[1][0][0] };
    const unsigned vlOff[2] = { (unsigned)(uintptr_t)&Vl[0][0][0],
                                (unsigned)(uintptr_t)&Vl[1][0][0] };

    // Stage one 32-key tile: K tile = contiguous 4 KB; V^T tile = 64 rows x 64 B.
    // 128 threads x 2 chunks x 16 B each => 4 async instructions per wave.
    auto stage = [&](int buf, int k0) {
        const char* kg = (const char*)kbh + (size_t)k0 * (DD * 2);
#pragma unroll
        for (int s = 0; s < 2; ++s) {
            const int c = tid + s * 128;          // chunk id 0..255
            async_cp16(klOff[buf] + c * 16, kg + (size_t)c * 16);
        }
        const char* vg = (const char*)vbh + (size_t)k0 * 2;
#pragma unroll
        for (int s = 0; s < 2; ++s) {
            const int c = tid + s * 128;          // chunk id 0..255
            const int row = c >> 2, col16 = c & 3;
            async_cp16(vlOff[buf] + c * 16,
                       vg + (size_t)row * (NN * 2) + col16 * 16);
        }
    };

    const int iters = NN / 32;
    stage(0, 0);
    for (int it = 0; it < iters; ++it) {
        const int buf = it & 1;
        const int k0  = it * 32;
        if (it + 1 < iters) {
            stage(buf ^ 1, k0 + 32);
            asm volatile("s_wait_asynccnt 0x4" ::: "memory");  // current tile done
        } else {
            asm volatile("s_wait_asynccnt 0x0" ::: "memory");
        }
        __syncthreads();   // all waves' staging visible

        // ---- scores: S[16q x 32k], two 16x16 C-frags, K=64 contraction ----
        v8f s[2];
#pragma unroll
        for (int t = 0; t < 2; ++t) {
            const unsigned short* krow = &Kl[buf][t * 16 + nidx][0];
            Frag16 b0, b1;
#pragma unroll
            for (int j = 0; j < 8; ++j) {
                b0.u[j] = *(const unsigned*)(krow + half * 16 + 2 * j);
                b1.u[j] = *(const unsigned*)(krow + 32 + half * 16 + 2 * j);
            }
            v8f acc = {};
            acc = __builtin_amdgcn_wmma_f32_16x16x32_bf16(false, qa[0].v, false, b0.v,
                                                          (short)0, acc, false, false);
            acc = __builtin_amdgcn_wmma_f32_16x16x32_bf16(false, qa[1].v, false, b1.v,
                                                          (short)0, acc, false, false);
            s[t] = acc;
        }

        // ---- online softmax (row = (r, half); 16 lanes per row group) ----
#pragma unroll
        for (int r = 0; r < 8; ++r) {
            float v0 = s[0][r] * scale;
            float v1 = s[1][r] * scale;
            float mx = fmaxf(v0, v1);
#pragma unroll
            for (int off = 8; off >= 1; off >>= 1)
                mx = fmaxf(mx, __shfl_xor(mx, off, 32));
            const float mn    = fmaxf(m[r], mx);
            const float alpha = __expf(m[r] - mn);
            m[r] = mn;
            const float p0 = __expf(v0 - mn);
            const float p1 = __expf(v1 - mn);
            float ps = p0 + p1;
#pragma unroll
            for (int off = 8; off >= 1; off >>= 1)
                ps += __shfl_xor(ps, off, 32);
            l[r] = l[r] * alpha + ps;
#pragma unroll
            for (int c = 0; c < 4; ++c) oacc[c][r] *= alpha;
            const int Mr = r + half * 8;
            Pl[wave][Mr][nidx]      = f2bf(p0);
            Pl[wave][Mr][nidx + 16] = f2bf(p1);
        }
        wave_lds_fence();  // per-wave slice: HW DS ordering suffices

        // ---- P (C layout) -> A-fragment via LDS gather ----
        Frag16 pa;
#pragma unroll
        for (int j = 0; j < 8; ++j) {
            int k = half * 16 + (j << 1);
            pa.u[j] = *(const unsigned*)(&Pl[wave][nidx][k]);
        }

        // ---- O += P(16x32) x V(32x64): 4 WMMAs over D chunks ----
#pragma unroll
        for (int c = 0; c < 4; ++c) {
            const unsigned short* vrow = &Vl[buf][c * 16 + nidx][0];
            Frag16 vb;
#pragma unroll
            for (int j = 0; j < 8; ++j)
                vb.u[j] = *(const unsigned*)(vrow + half * 16 + 2 * j);
            oacc[c] = __builtin_amdgcn_wmma_f32_16x16x32_bf16(false, pa.v, false, vb.v,
                                                              (short)0, oacc[c], false, false);
        }
        __syncthreads();   // protect K/V buffer before next overwrite
    }

    // ---- finalize: O /= l, write fp32 in "b n (h d)" layout ----
    const int h = bh % HH;
    const int b = bh / HH;
#pragma unroll
    for (int r = 0; r < 8; ++r) {
        const float inv = 1.0f / l[r];
        const int row = qbase + r + half * 8;
        float* orow = o + ((size_t)b * NN + row) * DM + h * DD;
#pragma unroll
        for (int c = 0; c < 4; ++c)
            orow[c * 16 + nidx] = oacc[c][r] * inv;
    }
}

// ---------------------------------------------------------------------------
// Kernel 3: output projection Y = O[B*N,512] x Wp^T + bp.
// One wave = one 16x16 output tile, 16 chained bf16 WMMAs (K=512).
// ---------------------------------------------------------------------------
__global__ __launch_bounds__(256) void oproj_kernel(
    const float* __restrict__ o,
    const float* __restrict__ Wp, const float* __restrict__ bp,
    float* __restrict__ y)
{
    const int lane = threadIdx.x & 31;
    const int wave = threadIdx.x >> 5;
    const int tile = blockIdx.x * 8 + wave;    // (B*N/16) * (DM/16) tiles
    const int CT   = DM / 16;
    const int rt   = tile / CT;
    const int ct   = tile % CT;

    const int nidx = lane & 15;
    const int half = lane >> 4;

    const float* arow = o  + ((size_t)rt * 16 + nidx) * DM;
    const float* wrow = Wp + ((size_t)ct * 16 + nidx) * DM;

    v8f acc = {};
    for (int kc = 0; kc < DM / 32; ++kc) {
        Frag16 a, bw;
#pragma unroll
        for (int j = 0; j < 8; ++j) {
            int ka = kc * 32 + half * 16 + (j << 1);
            a.u[j] = pack2(arow[ka], arow[ka + 1]);
            int kk = kc * 32 + half * 16 + (j << 1);
            bw.u[j] = pack2(wrow[kk], wrow[kk + 1]);
        }
        acc = __builtin_amdgcn_wmma_f32_16x16x32_bf16(false, a.v, false, bw.v,
                                                      (short)0, acc, false, false);
    }

    const float bias = bp[ct * 16 + nidx];
#pragma unroll
    for (int r = 0; r < 8; ++r) {
        const size_t row = (size_t)rt * 16 + r + half * 8;
        y[row * DM + ct * 16 + nidx] = acc[r] + bias;
    }
}

// ---------------------------------------------------------------------------
extern "C" void kernel_launch(void* const* d_in, const int* in_sizes, int n_in,
                              void* d_out, int out_size, void* d_ws, size_t ws_size,
                              hipStream_t stream) {
    const float* x  = (const float*)d_in[0];
    const float* Wq = (const float*)d_in[1];
    const float* bq = (const float*)d_in[2];
    const float* Wk = (const float*)d_in[3];
    const float* bk = (const float*)d_in[4];
    const float* Wv = (const float*)d_in[5];
    const float* bv = (const float*)d_in[6];
    const float* Wp = (const float*)d_in[7];
    const float* bp = (const float*)d_in[8];
    float* y = (float*)d_out;

    char* ws = (char*)d_ws;
    const size_t qkvBytes = (size_t)BB * HH * NN * DD * sizeof(unsigned short); // 16 MB each
    unsigned short* qb = (unsigned short*)(ws);
    unsigned short* kb = (unsigned short*)(ws + qkvBytes);
    unsigned short* vt = (unsigned short*)(ws + 2 * qkvBytes);
    float*          oo = (float*)        (ws + 3 * qkvBytes);                   // 33.5 MB

    const int tiles = BB * HH * (NN / 16);          // 8192
    qkv_proj_kernel<<<tiles / 8, 256, 0, stream>>>(x, Wq, bq, Wk, bk, Wv, bv, qb, kb, vt);

    const int ablocks = BB * HH * (NN / 64);        // 2048 blocks x 4 waves
    flash_attn_kernel<<<ablocks, 128, 0, stream>>>(qb, kb, vt, oo);

    const int otiles = (BB * NN / 16) * (DM / 16);  // 32768
    oproj_kernel<<<otiles / 8, 256, 0, stream>>>(oo, Wp, bp, y);
}
